// CRF_27831388078553
// MI455X (gfx1250) — matrix operand order repaired and checked by reading, not compiled, not run on hardware
//
#include <hip/hip_runtime.h>

// CRF log-likelihood on MI455X (gfx1250):
//  - gold-path score: trivial gather/sum kernel
//  - forward scan (dominant): exp-domain reformulation so each step is a
//    16x256x256 f16 GEMM per workgroup -> v_wmma_f32_16x16x32_f16, with
//    exp(transitions) held RESIDENT in VGPRs as WMMA B-fragments, alpha/p
//    kept in LDS in A-fragment layout, per-row rescaling for stability.
//    Emissions stream into LDS via double-buffered
//    global_load_async_to_lds_b128 (ASYNCcnt pipeline, branch-free).
//  - finalize: mean(score - logZ) over 64 batches.

typedef __attribute__((ext_vector_type(16))) _Float16 v16h;
typedef __attribute__((ext_vector_type(8)))  float    v8f;

constexpr int Bb  = 64;    // batch
constexpr int S   = 2048;  // sequence length
constexpr int T   = 256;   // tags / states
constexpr int WGB = 16;    // batches per workgroup (WMMA M dimension)
constexpr int NW  = 8;     // waves per workgroup (each owns 32 output states)
constexpr int EST = 260;   // padded emit row stride (floats); row = 1040 B (16B aligned)

// ---------------------------------------------------------------- score ----
__global__ void crf_score_kernel(const float* __restrict__ emissions,
                                 const int*   __restrict__ tags,
                                 const int*   __restrict__ mask,
                                 const float* __restrict__ trans,
                                 const float* __restrict__ startT,
                                 const float* __restrict__ endT,
                                 float*       __restrict__ scores) {
  const int b   = blockIdx.x;
  const int tid = threadIdx.x;
  __shared__ float sred[256];
  __shared__ int   mred[256];
  float part = 0.f;
  int   msum = 0;
  for (int s = tid; s < S; s += 256) {
    const int cur = tags[b * S + s];
    msum += mask[b * S + s];
    if (s == 0) {
      part += startT[cur] + emissions[(size_t)b * S * T + cur];
    } else {
      const int prev = tags[b * S + s - 1];
      // reference: transitions[tags[1:], tags[:-1]]  ->  trans[cur, prev]
      const float tr = trans[cur * T + prev];
      const float em = emissions[((size_t)b * S + s) * T + cur];
      part += (tr + em) * (float)mask[b * S + s];
    }
  }
  sred[tid] = part;
  mred[tid] = msum;
  __syncthreads();
  for (int off = 128; off > 0; off >>= 1) {
    if (tid < off) { sred[tid] += sred[tid + off]; mred[tid] += mred[tid + off]; }
    __syncthreads();
  }
  if (tid == 0) {
    const int last = tags[b * S + (mred[0] - 1)];
    scores[b] = sred[0] + endT[last];
  }
}

// -------------------------------------------------------------- forward ----
__global__ void __launch_bounds__(256, 1)
crf_forward_kernel(const float* __restrict__ emissions,
                   const float* __restrict__ trans,
                   const float* __restrict__ startT,
                   const float* __restrict__ endT,
                   float*       __restrict__ logz) {
  const int tid  = threadIdx.x;
  const int lane = tid & 31;
  const int w    = tid >> 5;     // wave id 0..7
  const int hi   = lane >> 4;    // half-wave
  const int ln   = lane & 15;
  const int B0   = blockIdx.x * WGB;

  // P (normalized alpha, f16) kept in A-fragment layout: [kblock][lane][elem]
  __shared__ __align__(32) _Float16 Pfrag[8][32][16];     // 8 KB
  __shared__ __align__(16) float ebuf[2][WGB][EST];       // 33 KB raw-emission double buffer
  __shared__ __align__(16) float rowred[WGB][NW];         // per-row per-wave reduction slots
  __shared__ float lsc[WGB];                              // per-row accumulated log-scale

  // Async stage of step tt's 16x256 f32 emission block into ebuf[nb]:
  // 4 x global_load_async_to_lds_b128 per thread, coalesced 16B per lane.
  auto issue_async = [&](int tt, int nb) {
    #pragma unroll
    for (int p = 0; p < 4; ++p) {
      const int idx4 = p * 1024 + tid * 4;    // float index in 16x256 block
      const int rr   = idx4 >> 8;
      const int c    = idx4 & 255;
      const unsigned la = (unsigned)(uintptr_t)(&ebuf[nb][rr][c]);   // LDS byte offset
      const unsigned go = (unsigned)((((B0 + rr) * S + tt) * T + c) * 4);
      asm volatile("global_load_async_to_lds_b128 %0, %1, %2"
                   :: "v"(la), "v"(go), "s"(emissions) : "memory");
    }
  };

  // Resident B operands: exp(transitions) fragments for this wave's 2 N-tiles.
  // B 32x16 layout: lanes 0-15 hold K=0..7,16..23; lanes 16-31 hold K=8..15,24..31 (matches A).
  v16h bA[8], bB[8];
  #pragma unroll
  for (int kb = 0; kb < 8; ++kb) {
    #pragma unroll
    for (int e = 0; e < 16; ++e) {
      const int k  = kb * 32 + hi * 8 + (e & 7) + ((e & 8) << 1);      // previous-state index (K dim)
      const int n0 = (2 * w + 0) * 16 + ln;      // next-state index (N dim), tile 0
      const int n1 = (2 * w + 1) * 16 + ln;      // tile 1
      bA[kb][e] = (_Float16)__expf(trans[k * T + n0]);
      bB[kb][e] = (_Float16)__expf(trans[k * T + n1]);
    }
  }

  const int j0 = w * 32 + ln;   // this lane's output state, tile 0
  const int j1 = j0 + 16;       // tile 1

  float q0[8], q1[8], gm[8];    // survive the loop for the final logZ stage

  // A-fragment scatter address for (row m, state k):
  //   lane = m + 16*((k>>3)&1), elem = (k&7) | ((k&16)>>1)   (within 32-wide K block)
  auto pstore = [&](int m, int k, float p) {
    const int kb = k >> 5, kib = k & 31;
    Pfrag[kb][m + ((kib >> 3) & 1) * 16][(kib & 7) | ((kib & 16) >> 1)] = (_Float16)p;
  };

  // ---- t == 0: alpha0 = start + emissions[:,0]; initialize scale + P ----
  {
    #pragma unroll 4
    for (int rr = 0; rr < WGB; ++rr)
      ebuf[0][rr][tid] = emissions[((size_t)(B0 + rr) * S) * T + tid] + startT[tid];
    issue_async(1, 1);                  // prefetch t=1 while we initialize
    __syncthreads();

    float rm[8];
    #pragma unroll
    for (int r = 0; r < 8; ++r) {
      const int m = r + 8 * hi;
      q0[r] = ebuf[0][m][j0];
      q1[r] = ebuf[0][m][j1];
      rm[r] = fmaxf(q0[r], q1[r]);
    }
    #pragma unroll
    for (int r = 0; r < 8; ++r) {
      rm[r] = fmaxf(rm[r], __shfl_xor(rm[r], 1, 16));
      rm[r] = fmaxf(rm[r], __shfl_xor(rm[r], 2, 16));
      rm[r] = fmaxf(rm[r], __shfl_xor(rm[r], 4, 16));
      rm[r] = fmaxf(rm[r], __shfl_xor(rm[r], 8, 16));
    }
    if (ln == 0) {
      #pragma unroll
      for (int r = 0; r < 8; ++r) rowred[r + 8 * hi][w] = rm[r];
    }
    __syncthreads();
    #pragma unroll
    for (int r = 0; r < 8; ++r) {
      const int m = r + 8 * hi;
      const float4* rp = (const float4*)(&rowred[m][0]);
      const float4 ga = rp[0], gb = rp[1];
      gm[r] = fmaxf(fmaxf(fmaxf(ga.x, ga.y), fmaxf(ga.z, ga.w)),
                    fmaxf(fmaxf(gb.x, gb.y), fmaxf(gb.z, gb.w)));
    }
    if (w == 0 && ln == 0) {
      #pragma unroll
      for (int r = 0; r < 8; ++r) lsc[r + 8 * hi] = gm[r];
    }
    #pragma unroll
    for (int r = 0; r < 8; ++r) {
      const int m = r + 8 * hi;
      pstore(m, j0, __expf(q0[r] - gm[r]));
      pstore(m, j1, __expf(q1[r] - gm[r]));
    }
    __syncthreads();
  }

  // ---- main scan: each step = 16x256x256 f16 WMMA GEMM + rescale ----
  for (int t = 1; t < S; ++t) {
    const int cur = t & 1;
    // Branch-free streaming: always issue the next block (tail issues a
    // redundant re-load of the last block into the dead buffer).
    const int tt = (t + 1 < S) ? (t + 1) : (S - 1);
    issue_async(tt, cur ^ 1);

    // Group all 16 ds_load_b128 A-fragment loads, then fence so they can't
    // sink; the 16 WMMAs then issue back-to-back behind one s_wait_dscnt.
    v16h a[8];
    #pragma unroll
    for (int kb = 0; kb < 8; ++kb)
      a[kb] = *(const v16h*)(&Pfrag[kb][lane][0]);
    asm volatile("" ::: "memory");      // scheduling fence: keep loads grouped

    v8f acc0 = {}, acc1 = {};
    #pragma unroll
    for (int kb = 0; kb < 8; ++kb) {
      acc0 = __builtin_amdgcn_wmma_f32_16x16x32_f16(false, a[kb], false, bA[kb],
                                                    (short)0, acc0, false, false);
      acc1 = __builtin_amdgcn_wmma_f32_16x16x32_f16(false, a[kb], false, bB[kb],
                                                    (short)0, acc1, false, false);
    }

    // This step's emission block (issued last iteration) is the older group
    // of 4 in-order async loads; the 4 just issued may stay in flight.
    asm volatile("s_wait_asynccnt 4" ::: "memory");
    __syncthreads();   // emissions visible; all waves' Pfrag reads complete

    // q = (P x expT) * exp(emit); per-row max for rescaling.
    float rm[8];
    #pragma unroll
    for (int r = 0; r < 8; ++r) {
      const int m = r + 8 * hi;
      q0[r] = acc0[r] * __expf(ebuf[cur][m][j0]);
      q1[r] = acc1[r] * __expf(ebuf[cur][m][j1]);
      rm[r] = fmaxf(q0[r], q1[r]);
    }
    #pragma unroll
    for (int r = 0; r < 8; ++r) {
      rm[r] = fmaxf(rm[r], __shfl_xor(rm[r], 1, 16));
      rm[r] = fmaxf(rm[r], __shfl_xor(rm[r], 2, 16));
      rm[r] = fmaxf(rm[r], __shfl_xor(rm[r], 4, 16));
      rm[r] = fmaxf(rm[r], __shfl_xor(rm[r], 8, 16));
    }
    if (ln == 0) {
      #pragma unroll
      for (int r = 0; r < 8; ++r) rowred[r + 8 * hi][w] = rm[r];
    }
    __syncthreads();

    // Cross-wave row max (vector ds reads); accumulate log-scale.
    #pragma unroll
    for (int r = 0; r < 8; ++r) {
      const int m = r + 8 * hi;
      const float4* rp = (const float4*)(&rowred[m][0]);
      const float4 ga = rp[0], gb = rp[1];
      gm[r] = fmaxf(fmaxf(fmaxf(ga.x, ga.y), fmaxf(ga.z, ga.w)),
                    fmaxf(fmaxf(gb.x, gb.y), fmaxf(gb.z, gb.w)));
    }
    if (w == 0 && ln == 0) {
      #pragma unroll
      for (int r = 0; r < 8; ++r) {
        const int m = r + 8 * hi;
        lsc[m] = lsc[m] + logf(gm[r]);
      }
    }

    // Normalize and scatter p back into A-fragment layout for the next step.
    #pragma unroll
    for (int r = 0; r < 8; ++r) {
      const int m = r + 8 * hi;
      const float inv = 1.0f / gm[r];
      pstore(m, j0, q0[r] * inv);
      pstore(m, j1, q1[r] * inv);
    }
    __syncthreads();
  }

  // logZ = lsc + log( sum_j p[j] * exp(end[j]) ), using last step's q/gm.
  float part[8];
  #pragma unroll
  for (int r = 0; r < 8; ++r) {
    const float inv = 1.0f / gm[r];
    part[r] = q0[r] * inv * __expf(endT[j0]) + q1[r] * inv * __expf(endT[j1]);
  }
  #pragma unroll
  for (int r = 0; r < 8; ++r) {
    part[r] += __shfl_xor(part[r], 1, 16);
    part[r] += __shfl_xor(part[r], 2, 16);
    part[r] += __shfl_xor(part[r], 4, 16);
    part[r] += __shfl_xor(part[r], 8, 16);
  }
  if (ln == 0) {
    #pragma unroll
    for (int r = 0; r < 8; ++r) rowred[r + 8 * hi][w] = part[r];
  }
  __syncthreads();
  if (tid < WGB) {
    const float4* rp = (const float4*)(&rowred[tid][0]);
    const float4 sa = rp[0], sb = rp[1];
    const float s = (sa.x + sa.y + sa.z + sa.w) + (sb.x + sb.y + sb.z + sb.w);
    logz[B0 + tid] = lsc[tid] + logf(s);
  }
}

// ------------------------------------------------------------- finalize ----
__global__ void crf_finalize_kernel(const float* __restrict__ scores,
                                    const float* __restrict__ logz,
                                    float* __restrict__ out) {
  __shared__ float sred[64];
  const int tid = threadIdx.x;
  sred[tid] = scores[tid] - logz[tid];
  __syncthreads();
  for (int off = 32; off > 0; off >>= 1) {
    if (tid < off) sred[tid] += sred[tid + off];
    __syncthreads();
  }
  if (tid == 0) out[0] = sred[0] * (1.0f / (float)Bb);
}

// ---------------------------------------------------------------- entry ----
extern "C" void kernel_launch(void* const* d_in, const int* in_sizes, int n_in,
                              void* d_out, int out_size, void* d_ws, size_t ws_size,
                              hipStream_t stream) {
  (void)in_sizes; (void)n_in; (void)out_size; (void)ws_size;
  const float* emissions = (const float*)d_in[0];
  const int*   tags      = (const int*)d_in[1];
  const int*   mask      = (const int*)d_in[2];
  const float* trans     = (const float*)d_in[3];
  const float* startT    = (const float*)d_in[4];
  const float* endT      = (const float*)d_in[5];
  float* out    = (float*)d_out;
  float* scores = (float*)d_ws;      // 64 floats
  float* logz   = scores + 64;       // 64 floats

  crf_score_kernel<<<Bb, 256, 0, stream>>>(emissions, tags, mask, trans,
                                           startT, endT, scores);
  crf_forward_kernel<<<Bb / WGB, 256, 0, stream>>>(emissions, trans,
                                                   startT, endT, logz);
  crf_finalize_kernel<<<1, 64, 0, stream>>>(scores, logz, out);
}
